// RPN_3856880632072
// MI455X (gfx1250) — matrix operand order repaired and checked by reading, not compiled, not run on hardware
//
#include <hip/hip_runtime.h>

// ---------------------------------------------------------------------------
// Types for WMMA (gfx1250, wave32)
// ---------------------------------------------------------------------------
typedef __attribute__((ext_vector_type(16))) __bf16        v16bf;
typedef __attribute__((ext_vector_type(8)))  float         v8f;
typedef __attribute__((ext_vector_type(4)))  unsigned int  u32x4;
typedef __attribute__((ext_vector_type(4)))  int           i32x4;

union FragBF { u32x4 u[2]; v16bf v; };

__device__ __forceinline__ v8f wmma_bf16(const FragBF& a, const FragBF& b, v8f c) {
  // D = A(16x32 bf16) * B(32x16 bf16) + C(16x16 f32)
  return __builtin_amdgcn_wmma_f32_16x16x32_bf16(false, a.v, false, b.v,
                                                 (short)0, c, false, false);
}

// f32 -> bf16 round-to-nearest-even (bit level)
__device__ __forceinline__ unsigned short f2bf(float f) {
  unsigned u = __float_as_uint(f);
  unsigned r = 0x7fffu + ((u >> 16) & 1u);
  return (unsigned short)((u + r) >> 16);
}

// CDNA5 async global->LDS path (ASYNCcnt), guarded so we fall back cleanly.
#if defined(__has_builtin)
#if __has_builtin(__builtin_amdgcn_global_load_async_to_lds_b128) && \
    __has_builtin(__builtin_amdgcn_s_wait_asynccnt)
#define USE_ASYNC_LDS 1
#endif
#endif
#ifndef USE_ASYNC_LDS
#define USE_ASYNC_LDS 0
#endif

#if USE_ASYNC_LDS
// Signature (from hipcc diagnostic): param0 = int4 AS(1)*, param1 = int4 AS(3)*,
// then imm offset, imm cpol.
__device__ __forceinline__ void async_g2l_b128(const void* g, void* l) {
  __builtin_amdgcn_global_load_async_to_lds_b128(
      (__attribute__((address_space(1))) i32x4*)(g),
      (__attribute__((address_space(3))) i32x4*)(l), 0, 0);
}
#endif

// Problem constants
#define BATCH   4
#define CIN     256
#define HH      128
#define WW      128
#define HP      130         /* padded spatial dim (1-px zero border) */
#define NPI     147456      /* H*W*A per image  */
#define MTOT    65536       /* B*H*W            */
#define PRE_K   600
#define POST_K  100

// ---------------------------------------------------------------------------
// Prep kernels
// ---------------------------------------------------------------------------
// features NCHW f32 -> zero-padded NHWC bf16 [4][130][130][256]
__global__ void k_prep_feat(const float* __restrict__ in,
                            unsigned short* __restrict__ out, int n) {
  int i = blockIdx.x * blockDim.x + threadIdx.x;
  if (i >= n) return;
  int c  = i & 255;
  int r  = i >> 8;          // b*130*130 + py*130 + px
  int px = r % HP;
  int r2 = r / HP;
  int py = r2 % HP;
  int b  = r2 / HP;
  unsigned short v = 0;
  if (px >= 1 && px <= WW && py >= 1 && py <= HH)
    v = f2bf(in[(((size_t)b * CIN + c) * HH + (py - 1)) * WW + (px - 1)]);
  out[i] = v;
}

// conv_w (cout,cin,3,3) f32 -> Wt[cout][k] bf16 with k=(ky*3+kx)*256+cin
__global__ void k_prep_w1(const float* __restrict__ w,
                          unsigned short* __restrict__ out, int n) {
  int i = blockIdx.x * blockDim.x + threadIdx.x;
  if (i >= n) return;
  int k = i % 2304, cout = i / 2304;
  int cin = k & 255, kk = k >> 8;
  int ky = kk / 3, kx = kk % 3;
  out[i] = f2bf(w[(((size_t)cout * CIN + cin) * 3 + ky) * 3 + kx]);
}

// merged head weights [48][256] bf16 (9 cls + 36 bbox + 3 zero pad) + bias[48]
__global__ void k_prep_whead(const float* __restrict__ cw, const float* __restrict__ bw,
                             const float* __restrict__ cb, const float* __restrict__ bb,
                             unsigned short* __restrict__ wh, float* __restrict__ hb) {
  int i = blockIdx.x * blockDim.x + threadIdx.x;
  if (i < 48 * 256) {
    int n = i >> 8, c = i & 255;
    float v = 0.f;
    if (n < 9)       v = cw[n * 256 + c];
    else if (n < 45) v = bw[(n - 9) * 256 + c];
    wh[i] = f2bf(v);
  }
  if (i < 48) {
    float v = 0.f;
    if (i < 9)       v = cb[i];
    else if (i < 45) v = bb[i - 9];
    hb[i] = v;
  }
}

// ---------------------------------------------------------------------------
// 3x3 conv as implicit GEMM: M=65536 (b,y,x), N=256 (cout), K=2304 (ky,kx,cin)
// Workgroup: 256 thr (8 waves). Tile: M=64 (x-strip), N=256. K step = 32.
// Double-buffered LDS; async global->LDS when available (uniform 5 async
// instructions per wave per stage -> deterministic s_wait_asynccnt values).
// LDS row stride 80B -> conflict-free b128 ds reads, 16B aligned.
// ---------------------------------------------------------------------------
__global__ __launch_bounds__(256) void k_conv3x3(
    const unsigned short* __restrict__ featp,  // padded NHWC bf16 [4][130][130][256]
    const unsigned short* __restrict__ wt,     // [256][2304] bf16
    const float* __restrict__ conv_b,
    unsigned short* __restrict__ t)            // NHWC bf16 out [4][128][128][256]
{
  __shared__ char As[2][64 * 80];    // 64 pixels x 32 ch bf16
  __shared__ char Bs[2][256 * 80];   // 256 cout  x 32 k  bf16

  const int tid  = threadIdx.x;
  const int bw   = blockIdx.x;          // 4*128*2 = 1024 blocks
  const int x0   = (bw & 1) * 64;
  const int y    = (bw >> 1) & 127;
  const int b    = bw >> 8;
  const int lane = tid & 31;
  const int wv   = tid >> 5;            // wave 0..7
  const int mi   = wv & 3;              // M sub-block (16 rows)
  const int ni   = wv >> 2;             // N half (128 couts)
  const int arow = tid >> 2, achunk = tid & 3;

  v8f acc[8] = {};

  // Stage loader: 1 A chunk + 4 B chunks per thread (always in-bounds: padded).
  auto issue = [&](int s, int buf) {
    const int kk = s >> 3, cc = s & 7;
    const int ky = kk / 3, kx = kk % 3;
    const int c0 = cc * 32;
    const unsigned short* asrc =
        featp + (((size_t)b * HP + (y + ky)) * HP + (x0 + arow + kx)) * 256
              + c0 + achunk * 8;
    char* adst = As[buf] + arow * 80 + achunk * 16;
    const int kbase = kk * 256 + c0 + achunk * 8;
#if USE_ASYNC_LDS
    async_g2l_b128(asrc, adst);
#pragma unroll
    for (int rr = 0; rr < 4; ++rr) {
      const int row = arow + rr * 64;
      async_g2l_b128(wt + (size_t)row * 2304 + kbase,
                     Bs[buf] + row * 80 + achunk * 16);
    }
#else
    *(u32x4*)adst = *(const u32x4*)asrc;
#pragma unroll
    for (int rr = 0; rr < 4; ++rr) {
      const int row = arow + rr * 64;
      *(u32x4*)(Bs[buf] + row * 80 + achunk * 16) =
          *(const u32x4*)(wt + (size_t)row * 2304 + kbase);
    }
#endif
  };

  issue(0, 0);
  for (int s = 0; s < 72; ++s) {
    const int buf = s & 1;
    if (s + 1 < 72) issue(s + 1, buf ^ 1);    // prefetch next stage
#if USE_ASYNC_LDS
    if (s + 1 < 72) __builtin_amdgcn_s_wait_asynccnt(5);  // stage s done; s+1 in flight
    else            __builtin_amdgcn_s_wait_asynccnt(0);
#endif
    __syncthreads();

    // A fragment: lane<16 -> row=lane, K{0..7,16..23}; lane>=16 -> K{8..15,24..31}
    FragBF afr;
    {
      const int row = mi * 16 + (lane & 15);
      const int kh  = (lane >> 4) * 16;
      afr.u[0] = *(const u32x4*)(As[buf] + row * 80 + kh);
      afr.u[1] = *(const u32x4*)(As[buf] + row * 80 + kh + 32);
    }
    // B fragments (lane<16 -> col=lane, K{0..7,16..23}; lane>=16 -> K{8..15,24..31}),
    // two-deep software pipeline so ds_loads overlap WMMA issue.
    const int kb = (lane >> 4) * 16;
    FragBF bfr[2];
    {
      const int brow = ni * 128 + (lane & 15);
      bfr[0].u[0] = *(const u32x4*)(Bs[buf] + brow * 80 + kb);
      bfr[0].u[1] = *(const u32x4*)(Bs[buf] + brow * 80 + kb + 32);
    }
#pragma unroll
    for (int nt = 0; nt < 8; ++nt) {
      if (nt + 1 < 8) {
        const int brow = ni * 128 + (nt + 1) * 16 + (lane & 15);
        bfr[(nt + 1) & 1].u[0] = *(const u32x4*)(Bs[buf] + brow * 80 + kb);
        bfr[(nt + 1) & 1].u[1] = *(const u32x4*)(Bs[buf] + brow * 80 + kb + 32);
      }
      acc[nt] = wmma_bf16(afr, bfr[nt & 1], acc[nt]);
    }
    __syncthreads();   // protect buf from next-next stage's refill
  }

  // Epilogue: bias + ReLU + bf16 store (C layout: vgpr r -> M=r(+8), lane -> N)
#pragma unroll
  for (int nt = 0; nt < 8; ++nt) {
    const int cout = ni * 128 + nt * 16 + (lane & 15);
    const float bias = conv_b[cout];
    const int mb = mi * 16 + (lane >> 4) * 8;
#pragma unroll
    for (int r = 0; r < 8; ++r) {
      const int x = x0 + mb + r;
      float v = acc[nt][r] + bias;
      v = v > 0.f ? v : 0.f;
      t[((((size_t)b * HH + y) * WW + x) << 8) + cout] = f2bf(v);
    }
  }
}

// ---------------------------------------------------------------------------
// Head 1x1 convs as one GEMM: M=65536, N=48 (9 cls + 36 bbox + 3 pad), K=256
// ---------------------------------------------------------------------------
__global__ __launch_bounds__(256) void k_head(
    const unsigned short* __restrict__ t,   // NHWC bf16
    const unsigned short* __restrict__ wh,  // [48][256] bf16
    const float* __restrict__ hb,           // [48]
    float* __restrict__ scores,             // [4][147456]
    float* __restrict__ deltas)             // [4][147456][4]
{
  __shared__ char As[128 * 80];
  __shared__ char Bs[48 * 80];
  const int tid = threadIdx.x;
  const int lane = tid & 31;
  const int wv = tid >> 5;
  const int m0 = blockIdx.x * 128;

  v8f acc[3] = {};
  for (int s = 0; s < 8; ++s) {
    const int c0 = s * 32;
    __syncthreads();
#pragma unroll
    for (int i = 0; i < 2; ++i) {
      const int idx = tid + i * 256;
      const int row = idx >> 2, chunk = idx & 3;
      *(u32x4*)(As + row * 80 + chunk * 16) =
          *(const u32x4*)(t + (((size_t)(m0 + row)) << 8) + c0 + chunk * 8);
    }
    if (tid < 192) {
      const int row = tid >> 2, chunk = tid & 3;
      *(u32x4*)(Bs + row * 80 + chunk * 16) =
          *(const u32x4*)(wh + ((size_t)row << 8) + c0 + chunk * 8);
    }
    __syncthreads();
    FragBF afr;
    {
      const int row = wv * 16 + (lane & 15);
      const int kh  = (lane >> 4) * 16;
      afr.u[0] = *(const u32x4*)(As + row * 80 + kh);
      afr.u[1] = *(const u32x4*)(As + row * 80 + kh + 32);
    }
#pragma unroll
    for (int nt = 0; nt < 3; ++nt) {
      const int row = nt * 16 + (lane & 15);
      const int kb  = (lane >> 4) * 16;
      FragBF bfr;
      bfr.u[0] = *(const u32x4*)(Bs + row * 80 + kb);
      bfr.u[1] = *(const u32x4*)(Bs + row * 80 + kb + 32);
      acc[nt] = wmma_bf16(afr, bfr, acc[nt]);
    }
  }
#pragma unroll
  for (int nt = 0; nt < 3; ++nt) {
    const int n = nt * 16 + (lane & 15);
    const float bias = hb[n];
    const int mb = wv * 16 + (lane >> 4) * 8;
#pragma unroll
    for (int r = 0; r < 8; ++r) {
      const int m = m0 + mb + r;
      const int b = m >> 14;         // /16384
      const int p = m & 16383;
      const float v = acc[nt][r] + bias;
      if (n < 9) {
        scores[(size_t)b * NPI + p * 9 + n] = v;            // [b][p*9+a]
      } else if (n < 45) {
        const int ch = n - 9;
        deltas[(((size_t)b * NPI) + (size_t)p * 9 + (ch >> 2)) * 4 + (ch & 3)] = v;
      }
    }
  }
}

// ---------------------------------------------------------------------------
// Box decode + clip (anchors computed analytically, scales outer/ratios inner)
// ---------------------------------------------------------------------------
__global__ void k_decode(const float* __restrict__ deltas, float* __restrict__ boxes,
                         const int* __restrict__ imh, const int* __restrict__ imw) {
  const int g = blockIdx.x * blockDim.x + threadIdx.x;
  if (g >= BATCH * NPI) return;
  const int i = g % NPI;
  const int a = i % 9;
  const int p = i / 9;
  const int x = p & 127, y = p >> 7;
  const int si = a / 3, ri = a % 3;
  const float scale = 8.f * (float)(1 << si);
  const float sq = (ri == 0) ? 0.70710678118654752440f
                             : ((ri == 1) ? 1.0f : 1.41421356237309504880f);
  const float w = 16.f * scale * sq;
  const float h = 16.f * scale / sq;
  const float gx = (float)x * 16.f;
  const float gy = (float)y * 16.f;
  const float dx = deltas[(size_t)g * 4 + 0];
  const float dy = deltas[(size_t)g * 4 + 1];
  const float dw = deltas[(size_t)g * 4 + 2];
  const float dh = deltas[(size_t)g * 4 + 3];
  const float pcx = gx + dx * w;
  const float pcy = gy + dy * h;
  const float pw = w * expf(dw);
  const float ph = h * expf(dh);
  const float Wc = (float)imw[0];
  const float Hc = (float)imh[0];
  float x1 = pcx - 0.5f * pw, y1 = pcy - 0.5f * ph;
  float x2 = pcx + 0.5f * pw, y2 = pcy + 0.5f * ph;
  x1 = fminf(fmaxf(x1, 0.f), Wc);
  y1 = fminf(fmaxf(y1, 0.f), Hc);
  x2 = fminf(fmaxf(x2, 0.f), Wc);
  y2 = fminf(fmaxf(y2, 0.f), Hc);
  boxes[(size_t)g * 4 + 0] = x1;
  boxes[(size_t)g * 4 + 1] = y1;
  boxes[(size_t)g * 4 + 2] = x2;
  boxes[(size_t)g * 4 + 3] = y2;
}

// ---------------------------------------------------------------------------
// Exact top-600 per image, descending, index-stable ties. Tournament selection.
// 147456 == 144 * 1024 exactly.
// ---------------------------------------------------------------------------
__global__ __launch_bounds__(1024) void k_topk(const float* __restrict__ scores,
                                               int* __restrict__ topi) {
  const int b = blockIdx.x;
  const float* s = scores + (size_t)b * NPI;
  __shared__ unsigned sel[NPI / 32];   // 4608 words = 18 KB
  __shared__ float rs[1024];
  __shared__ int   ri[1024];
  __shared__ int   winner;
  const int tid = threadIdx.x;
  for (int i = tid; i < NPI / 32; i += 1024) sel[i] = 0u;
  __syncthreads();

  float bestS = -3.402823466e38f;
  int   bestI = 0x7fffffff;
  for (int k = 0; k < 144; ++k) {
    const int idx = tid + (k << 10);
    const float v = s[idx];
    if (v > bestS || (v == bestS && idx < bestI)) { bestS = v; bestI = idx; }
  }

  for (int r = 0; r < PRE_K; ++r) {
    rs[tid] = bestS; ri[tid] = bestI;
    __syncthreads();
    for (int off = 512; off > 0; off >>= 1) {
      if (tid < off) {
        const float s2 = rs[tid + off];
        const int   i2 = ri[tid + off];
        if (s2 > rs[tid] || (s2 == rs[tid] && i2 < ri[tid])) { rs[tid] = s2; ri[tid] = i2; }
      }
      __syncthreads();
    }
    if (tid == 0) { winner = ri[0]; topi[b * PRE_K + r] = ri[0]; }
    __syncthreads();
    const int wi = winner;
    if ((wi & 1023) == tid) {          // owner thread marks + rescans its slice
      sel[wi >> 5] |= 1u << (wi & 31);
      bestS = -3.402823466e38f; bestI = 0x7fffffff;
      for (int k = 0; k < 144; ++k) {
        const int idx = tid + (k << 10);
        if ((sel[idx >> 5] >> (idx & 31)) & 1u) continue;
        const float v = s[idx];
        if (v > bestS || (v == bestS && idx < bestI)) { bestS = v; bestI = idx; }
      }
    }
    __syncthreads();
  }
}

// ---------------------------------------------------------------------------
// Greedy NMS over the 600 (score-ordered) boxes + ordered gather of 100
// ---------------------------------------------------------------------------
__global__ __launch_bounds__(1024) void k_nms(const float* __restrict__ boxes,
                                              const int* __restrict__ topi,
                                              float* __restrict__ out) {
  const int b = blockIdx.x;
  const int tid = threadIdx.x;
  __shared__ float bx[PRE_K][4];
  __shared__ float area[PRE_K];
  __shared__ int   keep[PRE_K];
  const float* pb = boxes + (size_t)b * NPI * 4;
  const int*   ti = topi + b * PRE_K;

  if (tid < PRE_K) {
    const int idx = ti[tid];
    const float x1 = pb[(size_t)idx * 4 + 0];
    const float y1 = pb[(size_t)idx * 4 + 1];
    const float x2 = pb[(size_t)idx * 4 + 2];
    const float y2 = pb[(size_t)idx * 4 + 3];
    bx[tid][0] = x1; bx[tid][1] = y1; bx[tid][2] = x2; bx[tid][3] = y2;
    area[tid] = (x2 - x1) * (y2 - y1);
    keep[tid] = 1;
  }
  __syncthreads();

  for (int i = 0; i < PRE_K; ++i) {
    if (keep[i] && tid < PRE_K && tid > i && keep[tid]) {
      const float xx1 = fmaxf(bx[i][0], bx[tid][0]);
      const float yy1 = fmaxf(bx[i][1], bx[tid][1]);
      const float xx2 = fminf(bx[i][2], bx[tid][2]);
      const float yy2 = fminf(bx[i][3], bx[tid][3]);
      const float inter = fmaxf(xx2 - xx1, 0.f) * fmaxf(yy2 - yy1, 0.f);
      const float iou = inter / (area[i] + area[tid] - inter);
      if (iou > 0.7f) keep[tid] = 0;
    }
    __syncthreads();
  }

  if (tid == 0) {
    int cnt = 0;
    for (int k = 0; k < PRE_K && cnt < POST_K; ++k) {
      if (keep[k]) {
        float* o = out + ((size_t)b * POST_K + cnt) * 4;
        o[0] = bx[k][0]; o[1] = bx[k][1]; o[2] = bx[k][2]; o[3] = bx[k][3];
        ++cnt;
      }
    }
    for (; cnt < POST_K; ++cnt) {
      float* o = out + ((size_t)b * POST_K + cnt) * 4;
      o[0] = 0.f; o[1] = 0.f; o[2] = 0.f; o[3] = 0.f;
    }
  }
}

// ---------------------------------------------------------------------------
// Launch
// ---------------------------------------------------------------------------
extern "C" void kernel_launch(void* const* d_in, const int* in_sizes, int n_in,
                              void* d_out, int out_size, void* d_ws, size_t ws_size,
                              hipStream_t stream) {
  (void)in_sizes; (void)n_in; (void)out_size; (void)ws_size;
  const float* feats  = (const float*)d_in[0];
  const float* conv_w = (const float*)d_in[1];
  const float* conv_b = (const float*)d_in[2];
  const float* cls_w  = (const float*)d_in[3];
  const float* cls_b  = (const float*)d_in[4];
  const float* bbox_w = (const float*)d_in[5];
  const float* bbox_b = (const float*)d_in[6];
  const int*   img_h  = (const int*)d_in[7];
  const int*   img_w  = (const int*)d_in[8];

  char* ws = (char*)d_ws;
  size_t off = 0;
  auto alloc = [&](size_t bytes) -> void* {
    void* p = ws + off;
    off += (bytes + 255) & ~(size_t)255;
    return p;
  };
  const size_t npad = (size_t)BATCH * HP * HP * CIN;           // padded feat elems
  unsigned short* featb = (unsigned short*)alloc(npad * 2);
  unsigned short* wt1   = (unsigned short*)alloc((size_t)256 * 2304 * 2);
  unsigned short* wh    = (unsigned short*)alloc((size_t)48 * 256 * 2);
  float*          hb    = (float*)alloc(48 * 4);
  unsigned short* tb    = (unsigned short*)alloc((size_t)BATCH * HH * WW * CIN * 2);
  float*          sc    = (float*)alloc((size_t)BATCH * NPI * 4);
  float*          dl    = (float*)alloc((size_t)BATCH * NPI * 16);
  float*          bxs   = (float*)alloc((size_t)BATCH * NPI * 16);
  int*            topi  = (int*)alloc((size_t)BATCH * PRE_K * 4);

  k_prep_feat <<<(int)(npad / 256), 256, 0, stream>>>(feats, featb, (int)npad);
  k_prep_w1   <<<(256 * 2304) / 256, 256, 0, stream>>>(conv_w, wt1, 256 * 2304);
  k_prep_whead<<<48, 256, 0, stream>>>(cls_w, bbox_w, cls_b, bbox_b, wh, hb);
  k_conv3x3   <<<BATCH * HH * (WW / 64), 256, 0, stream>>>(featb, wt1, conv_b, tb);
  k_head      <<<MTOT / 128, 256, 0, stream>>>(tb, wh, hb, sc, dl);
  k_decode    <<<(BATCH * NPI) / 256, 256, 0, stream>>>(dl, bxs, img_h, img_w);
  k_topk      <<<BATCH, 1024, 0, stream>>>(sc, topi);
  k_nms       <<<BATCH, 1024, 0, stream>>>(bxs, topi, (float*)d_out);
}
